// HGT_87471303950599
// MI455X (gfx1250) — matrix operand order repaired and checked by reading, not compiled, not run on hardware
//
#include <hip/hip_runtime.h>
#include <hip/hip_bf16.h>
#include <math.h>

// ---------------------------------------------------------------------------
// HGT forward for MI455X (gfx1250).  All dense matmuls go through one generic
// strided/batched WMMA GEMM kernel (v_wmma_f32_16x16x32_bf16, f32 accumulate).
// 4 waves per workgroup share the weight tile through LDS (staged directly in
// WMMA B-fragment order); segment-softmax attention over edges uses native
// f32 global atomics and a monotone-uint encoding for float atomic max.
// ---------------------------------------------------------------------------

typedef __attribute__((ext_vector_type(16))) __bf16 v16bf;
typedef __attribute__((ext_vector_type(8)))  float  v8f;

#define HID    256
#define NHEAD  4
#define HD     64
#define KQVLD  768   // fused [k|q|v] row width
#define QOFF   256   // q chunk offset within kqv row
#define VOFF   512   // v chunk offset within kqv row

#define GEMM_WAVES 4          // waves per workgroup; block tile = 64 x 16

__device__ __forceinline__ float gelu_exact(float x) {
    return 0.5f * x * (1.0f + erff(x * 0.70710678118654752f));
}

// monotone float<->uint encoding so unsigned atomicMax == float max
__device__ __forceinline__ unsigned fenc(float f) {
    unsigned u = __float_as_uint(f);
    return (u & 0x80000000u) ? ~u : (u | 0x80000000u);
}
__device__ __forceinline__ float fdec(unsigned u) {
    return __uint_as_float((u & 0x80000000u) ? (u & 0x7FFFFFFFu) : ~u);
}

// ---------------------------------------------------------------------------
// Generic GEMM: C[z] = (act(A[z]) @ W[z]) + bias, optional gated residual.
//   A: M x K, row stride lda, per-batch element offset aBatch
//   W: K x N, row stride ldw, per-batch element offset wBatch
//   C: M x N, row stride ldc, per-batch element offset cBatch
// 128 threads = 4 waves; wave w computes the 16x16 tile at rows
// blockIdx.x*64 + w*16, cols blockIdx.y*16.  The 32x16 B slab per K-step is
// cooperatively loaded (float4, coalesced), converted to bf16 and stored to
// LDS in WMMA B-fragment order; each lane then reads its 32-byte fragment
// with one vector ds_load.  WMMA layouts per CDNA5 ISA 7.12.2 (wave32, K=32):
//   A frag: lane M=lane&15; elems 0..7 -> K=8g+i, 8..15 -> K=16+8g+i (g=lane>>4)
//   B frag: lane N=lane&15; elems 0..7 -> K=8g+i, 8..15 -> K=16+8g+i (g=lane>>4)
//   C frag: vgpr r -> M=r+8g ; N=lane&15
// ---------------------------------------------------------------------------
__global__ __launch_bounds__(32 * GEMM_WAVES) void gemm_bf16_wmma(
    const float* __restrict__ A, long lda, long aBatch,
    const float* __restrict__ W, long ldw, long wBatch,
    const float* __restrict__ bias,
    const float* __restrict__ resid, long ldres,
    const float* __restrict__ skipLogit,
    float* __restrict__ C, long ldc, long cBatch,
    int M, int N, int K, int actGelu)
{
    (void)N;
    __shared__ __bf16 ldsB[32 * 16];   // [lane][elem], B fragment order, 1 KB

    const int tid  = threadIdx.x;      // 0..127
    const int lane = tid & 31;
    const int wave = tid >> 5;
    const int g    = lane >> 4;
    const int l15  = lane & 15;
    const int z    = blockIdx.z;
    const int m0   = blockIdx.x * (16 * GEMM_WAVES) + wave * 16;
    const int n0   = blockIdx.y * 16;

    const float* Ab = A + (size_t)z * aBatch;
    const float* Wb = W + (size_t)z * wBatch;
    float*       Cb = C + (size_t)z * cBatch;

    const int  arow  = m0 + l15;
    const bool rowOK = arow < M;
    const float* arp = Ab + (size_t)arow * lda;

    // cooperative B staging: thread t covers (row = t>>2, 4 cols at (t&3)*4)
    const int brow  = tid >> 2;          // 0..31  (K within step)
    const int bcol4 = (tid & 3) * 4;     // 0,4,8,12
    const int bg    = (brow >> 3) & 1;   // lane-half of this K row
    const int be    = (brow & 7) + ((brow >> 4) << 3);   // fragment element index

    v8f acc = {0.f, 0.f, 0.f, 0.f, 0.f, 0.f, 0.f, 0.f};

    for (int kb = 0; kb < K; kb += 32) {
        // ---- stage B slab into LDS in fragment order ----
        {
            const float4 w4 = *(const float4*)(Wb + (size_t)(kb + brow) * ldw + n0 + bcol4);
            ldsB[(size_t)((bcol4 + 0) + 16 * bg) * 16 + be] = (__bf16)w4.x;
            ldsB[(size_t)((bcol4 + 1) + 16 * bg) * 16 + be] = (__bf16)w4.y;
            ldsB[(size_t)((bcol4 + 2) + 16 * bg) * 16 + be] = (__bf16)w4.z;
            ldsB[(size_t)((bcol4 + 3) + 16 * bg) * 16 + be] = (__bf16)w4.w;
            if (kb + 32 < K)  // overlap next slab's L2 fetch with this WMMA
                __builtin_prefetch(Wb + (size_t)(kb + 32 + brow) * ldw + n0 + bcol4, 0, 1);
        }
        __syncthreads();

        // ---- A fragment (register-direct, two contiguous 32B runs) ----
        v16bf va;
        if (rowOK) {
            #pragma unroll
            for (int i = 0; i < 8; ++i) {
                float x0 = arp[kb + g * 8 + i];
                float x1 = arp[kb + 16 + g * 8 + i];
                if (actGelu) { x0 = gelu_exact(x0); x1 = gelu_exact(x1); }
                va[i]     = (__bf16)x0;
                va[8 + i] = (__bf16)x1;
            }
        } else {
            #pragma unroll
            for (int i = 0; i < 16; ++i) va[i] = (__bf16)0.f;
        }

        // ---- B fragment: one 32-byte vector ds_load ----
        const v16bf vb = *(const v16bf*)&ldsB[(size_t)lane * 16];

        acc = __builtin_amdgcn_wmma_f32_16x16x32_bf16(
                  false, va, false, vb, (short)0, acc, false, false);
        __syncthreads();   // protect LDS before next stage
    }

    float gate = 1.f;
    if (skipLogit) gate = 1.f / (1.f + expf(-skipLogit[0]));
    #pragma unroll
    for (int r = 0; r < 8; ++r) {
        const int mo = m0 + r + 8 * g;
        const int no = n0 + l15;
        if (mo < M) {
            float v = acc[r];
            if (bias)  v += bias[no];
            if (resid) v = gate * v + (1.f - gate) * resid[(size_t)mo * ldres + no];
            Cb[(size_t)mo * ldc + no] = v;
        }
    }
}

// ---------------------------------------------------------------------------
// Edge kernels (segment softmax attention + message scatter)
// ---------------------------------------------------------------------------
__global__ void fill_f32_kernel(float* p, float v, long n) {
    long i = (long)blockIdx.x * blockDim.x + threadIdx.x;
    if (i < n) p[i] = v;
}
__global__ void fill_u32_kernel(unsigned* p, unsigned v, long n) {
    long i = (long)blockIdx.x * blockDim.x + threadIdx.x;
    if (i < n) p[i] = v;
}

// logits[e,h] = scale * p_rel[h] * <krel[src[e],h,:], q[dst[e],h,:]> ; atomic max per (dst,h)
__global__ void edge_logits_kernel(const int* __restrict__ src, const int* __restrict__ dst,
                                   const float* __restrict__ krel, const float* __restrict__ kqvDst,
                                   const float* __restrict__ prel,
                                   float* __restrict__ logits, unsigned* __restrict__ mbuf, int E)
{
    long idx = (long)blockIdx.x * blockDim.x + threadIdx.x;
    if (idx >= (long)E * NHEAD) return;
    const int e = (int)(idx >> 2), h = (int)(idx & 3);
    const int si = src[e], di = dst[e];
    const float4* kr = (const float4*)(krel   + (size_t)si * HID + h * HD);
    const float4* qr = (const float4*)(kqvDst + (size_t)di * KQVLD + QOFF + h * HD);
    float acc = 0.f;
    #pragma unroll
    for (int d = 0; d < HD / 4; ++d) {
        float4 a = kr[d], b = qr[d];
        acc += a.x * b.x + a.y * b.y + a.z * b.z + a.w * b.w;
    }
    const float lg = acc * prel[h] * 0.125f;   // 1/sqrt(64)
    logits[idx] = lg;
    atomicMax(&mbuf[(size_t)di * NHEAD + h], fenc(lg));
}

// attn[e,h] = exp(logit - max) ; atomic sum per (dst,h)
__global__ void edge_expsum_kernel(const int* __restrict__ dst,
                                   const float* __restrict__ logits,
                                   const unsigned* __restrict__ mbuf,
                                   float* __restrict__ attn, float* __restrict__ sbuf, int E)
{
    long idx = (long)blockIdx.x * blockDim.x + threadIdx.x;
    if (idx >= (long)E * NHEAD) return;
    const int e = (int)(idx >> 2), h = (int)(idx & 3);
    const int di = dst[e];
    const float ex = __expf(logits[idx] - fdec(mbuf[(size_t)di * NHEAD + h]));
    attn[idx] = ex;
    __hip_atomic_fetch_add(&sbuf[(size_t)di * NHEAD + h], ex,
                           __ATOMIC_RELAXED, __HIP_MEMORY_SCOPE_AGENT);
}

__global__ void edge_norm_kernel(const int* __restrict__ dst,
                                 const float* __restrict__ sbuf,
                                 float* __restrict__ attn, int E)
{
    long idx = (long)blockIdx.x * blockDim.x + threadIdx.x;
    if (idx >= (long)E * NHEAD) return;
    const int e = (int)(idx >> 2), h = (int)(idx & 3);
    attn[idx] = attn[idx] / (sbuf[(size_t)dst[e] * NHEAD + h] + 1e-16f);
}

// agg[dst, c] += vrel[src, c] * attn[e, c/64]   (one thread per (edge, channel))
__global__ void edge_scatter_kernel(const int* __restrict__ src, const int* __restrict__ dst,
                                    const float* __restrict__ vrel, const float* __restrict__ attn,
                                    float* __restrict__ agg, int E)
{
    long idx = (long)blockIdx.x * blockDim.x + threadIdx.x;
    if (idx >= (long)E * HID) return;
    const int e = (int)(idx >> 8), c = (int)(idx & 255), h = c >> 6;
    const int si = src[e], di = dst[e];
    const float v = vrel[(size_t)si * HID + c] * attn[(size_t)e * NHEAD + h];
    __hip_atomic_fetch_add(&agg[(size_t)di * HID + c], v,
                           __ATOMIC_RELAXED, __HIP_MEMORY_SCOPE_AGENT);
}

// ---------------------------------------------------------------------------
// Host-side orchestration
// ---------------------------------------------------------------------------
static inline void gemm_go(hipStream_t s,
    const float* A, long lda, long aB,
    const float* W, long ldw, long wB,
    const float* bias, const float* resid, long ldres, const float* skip,
    float* C, long ldc, long cB, int M, int N, int K, int gelu, int batch)
{
    dim3 grid((unsigned)((M + 16 * GEMM_WAVES - 1) / (16 * GEMM_WAVES)),
              (unsigned)(N / 16), (unsigned)batch);
    gemm_bf16_wmma<<<grid, dim3(32 * GEMM_WAVES), 0, s>>>(
        A, lda, aB, W, ldw, wB, bias, resid, ldres, skip, C, ldc, cB, M, N, K, gelu);
}
static inline void fill_f32(hipStream_t s, float* p, float v, long n) {
    fill_f32_kernel<<<(unsigned)((n + 255) / 256), 256, 0, s>>>(p, v, n);
}
static inline void fill_u32(hipStream_t s, unsigned* p, unsigned v, long n) {
    fill_u32_kernel<<<(unsigned)((n + 255) / 256), 256, 0, s>>>(p, v, n);
}

extern "C" void kernel_launch(void* const* d_in, const int* in_sizes, int n_in,
                              void* d_out, int out_size, void* d_ws, size_t ws_size,
                              hipStream_t stream)
{
    (void)out_size; (void)ws_size;
    if (n_in < 42) return;

    const int IN_P = 128, IN_D = 64;
    const int NP = in_sizes[0] / IN_P;     // 50000
    const int ND = in_sizes[1] / IN_D;     // 5000
    const int E  = in_sizes[2];            // 400000

    const float* x_p      = (const float*)d_in[0];
    const float* x_d      = (const float*)d_in[1];
    const int*   tk_src   = (const int*)d_in[2];
    const int*   tk_dst   = (const int*)d_in[3];
    const int*   rv_src   = (const int*)d_in[4];
    const int*   rv_dst   = (const int*)d_in[5];

    // JAX tree-flatten (sorted dict keys): "layers" < "lin";
    // per layer: k_rel{rev,takes}, kqv{d:{W,b},p:{W,b}}, out{d,p}, p_rel{rev,takes},
    //            skip{d,p}, v_rel{rev,takes}  -> 16 leaves; lin{d:{W,b},p:{W,b}} last.
    const float* linW_d = (const float*)d_in[38];
    const float* linB_d = (const float*)d_in[39];
    const float* linW_p = (const float*)d_in[40];
    const float* linB_p = (const float*)d_in[41];

    // workspace carve (f32 elements)
    float* ws = (float*)d_ws;
    size_t off = 0;
    auto alloc = [&](size_t n) { float* p = ws + off; off += n; return p; };
    float* xpA   = alloc((size_t)NP * HID);
    float* xdA   = alloc((size_t)ND * HID);
    float* kqvP  = alloc((size_t)NP * KQVLD);
    float* kqvD  = alloc((size_t)ND * KQVLD);
    float* krelP = alloc((size_t)NP * HID);
    float* vrelP = alloc((size_t)NP * HID);
    float* krelD = alloc((size_t)ND * HID);
    float* vrelD = alloc((size_t)ND * HID);
    float* aggP  = alloc((size_t)NP * HID);
    float* aggD  = alloc((size_t)ND * HID);
    float* lgBuf = alloc((size_t)E * NHEAD);
    float* atBuf = alloc((size_t)E * NHEAD);
    unsigned* mbuf = (unsigned*)alloc((size_t)NP * NHEAD);
    float*    sbuf = alloc((size_t)NP * NHEAD);

    float* outP = (float*)d_out;
    float* outD = outP + (size_t)NP * HID;

    const unsigned eb = (unsigned)(((long)E * NHEAD + 255) / 256);
    const unsigned sb = (unsigned)(((long)E * HID + 255) / 256);

    // ---- input projections ----
    gemm_go(stream, x_p, IN_P, 0, linW_p, HID, 0, linB_p, nullptr, 0, nullptr,
            xpA, HID, 0, NP, HID, IN_P, 0, 1);
    gemm_go(stream, x_d, IN_D, 0, linW_d, HID, 0, linB_d, nullptr, 0, nullptr,
            xdA, HID, 0, ND, HID, IN_D, 0, 1);

    for (int L = 0; L < 2; ++L) {
        const int o = 6 + 16 * L;
        const float* krel_rev   = (const float*)d_in[o + 0];
        const float* krel_takes = (const float*)d_in[o + 1];
        const float* kqvW_d     = (const float*)d_in[o + 2];
        const float* kqvB_d     = (const float*)d_in[o + 3];
        const float* kqvW_p     = (const float*)d_in[o + 4];
        const float* kqvB_p     = (const float*)d_in[o + 5];
        const float* outW_d     = (const float*)d_in[o + 6];
        const float* outB_d     = (const float*)d_in[o + 7];
        const float* outW_p     = (const float*)d_in[o + 8];
        const float* outB_p     = (const float*)d_in[o + 9];
        const float* prel_rev   = (const float*)d_in[o + 10];
        const float* prel_takes = (const float*)d_in[o + 11];
        const float* skip_d     = (const float*)d_in[o + 12];
        const float* skip_p     = (const float*)d_in[o + 13];
        const float* vrel_rev   = (const float*)d_in[o + 14];
        const float* vrel_takes = (const float*)d_in[o + 15];

        // fused KQV projections
        gemm_go(stream, xpA, HID, 0, kqvW_p, KQVLD, 0, kqvB_p, nullptr, 0, nullptr,
                kqvP, KQVLD, 0, NP, KQVLD, HID, 0, 1);
        gemm_go(stream, xdA, HID, 0, kqvW_d, KQVLD, 0, kqvB_d, nullptr, 0, nullptr,
                kqvD, KQVLD, 0, ND, KQVLD, HID, 0, 1);

        // per-head relation transforms: y[n,h,:] = k[n,h,:] @ R[h]  (batched over grid.z)
        gemm_go(stream, kqvP + 0,    KQVLD, HD, krel_takes, HD, HD * HD, nullptr, nullptr, 0, nullptr,
                krelP, HID, HD, NP, HD, HD, 0, NHEAD);
        gemm_go(stream, kqvP + VOFF, KQVLD, HD, vrel_takes, HD, HD * HD, nullptr, nullptr, 0, nullptr,
                vrelP, HID, HD, NP, HD, HD, 0, NHEAD);
        gemm_go(stream, kqvD + 0,    KQVLD, HD, krel_rev,   HD, HD * HD, nullptr, nullptr, 0, nullptr,
                krelD, HID, HD, ND, HD, HD, 0, NHEAD);
        gemm_go(stream, kqvD + VOFF, KQVLD, HD, vrel_rev,   HD, HD * HD, nullptr, nullptr, 0, nullptr,
                vrelD, HID, HD, ND, HD, HD, 0, NHEAD);

        fill_f32(stream, aggP, 0.f, (long)NP * HID);
        fill_f32(stream, aggD, 0.f, (long)ND * HID);

        // relation: patient --takes--> drug  (dst = drug)
        fill_u32(stream, mbuf, 0u, (long)ND * NHEAD);
        fill_f32(stream, sbuf, 0.f, (long)ND * NHEAD);
        edge_logits_kernel<<<eb, 256, 0, stream>>>(tk_src, tk_dst, krelP, kqvD,
                                                   prel_takes, lgBuf, mbuf, E);
        edge_expsum_kernel<<<eb, 256, 0, stream>>>(tk_dst, lgBuf, mbuf, atBuf, sbuf, E);
        edge_norm_kernel<<<eb, 256, 0, stream>>>(tk_dst, sbuf, atBuf, E);
        edge_scatter_kernel<<<sb, 256, 0, stream>>>(tk_src, tk_dst, vrelP, atBuf, aggD, E);

        // relation: drug --rev_takes--> patient  (dst = patient)
        fill_u32(stream, mbuf, 0u, (long)NP * NHEAD);
        fill_f32(stream, sbuf, 0.f, (long)NP * NHEAD);
        edge_logits_kernel<<<eb, 256, 0, stream>>>(rv_src, rv_dst, krelD, kqvP,
                                                   prel_rev, lgBuf, mbuf, E);
        edge_expsum_kernel<<<eb, 256, 0, stream>>>(rv_dst, lgBuf, mbuf, atBuf, sbuf, E);
        edge_norm_kernel<<<eb, 256, 0, stream>>>(rv_dst, sbuf, atBuf, E);
        edge_scatter_kernel<<<sb, 256, 0, stream>>>(rv_src, rv_dst, vrelD, atBuf, aggP, E);

        // output projection: gelu(agg) @ W + b, sigmoid(skip)-gated residual.
        // Layer 0 writes in place (each element read-then-written by its owner
        // thread only); layer 1 writes straight into d_out.
        float* cP = (L == 0) ? xpA : outP;
        float* cD = (L == 0) ? xdA : outD;
        gemm_go(stream, aggP, HID, 0, outW_p, HID, 0, outB_p, xpA, HID, skip_p,
                cP, HID, 0, NP, HID, HID, 1, 1);
        gemm_go(stream, aggD, HID, 0, outW_d, HID, 0, outB_d, xdA, HID, skip_d,
                cD, HID, 0, ND, HID, HID, 1, 1);
    }
}
